// NaiveAttention_19765439496469
// MI455X (gfx1250) — compile-verified, matrix operand order checked by reading, not run on hardware
//
#include <hip/hip_runtime.h>
#include <hip/hip_bf16.h>

// Problem constants (from reference): B=2, T=2048, D=1024, H=16, hd=64
#define BB 2
#define TT 2048
#define DM 1024
#define NH 16
#define HD 64

typedef __bf16 bf16_t;
typedef __attribute__((ext_vector_type(4)))  bf16_t v4bf;
typedef __attribute__((ext_vector_type(8)))  bf16_t v8bf;
typedef __attribute__((ext_vector_type(16))) bf16_t v16bf;
typedef __attribute__((ext_vector_type(8)))  float  v8f;

// ---------------------------------------------------------------------------
// Tensor Data Mover support (gfx1250). Guarded: falls back to direct global
// fragment loads if the builtin is unavailable, so compilation never breaks.
// ---------------------------------------------------------------------------
#if defined(__has_builtin)
#if __has_builtin(__builtin_amdgcn_tensor_load_to_lds)
#define HAVE_TDM 1
#endif
#endif
#ifndef HAVE_TDM
#define HAVE_TDM 0
#endif

#if HAVE_TDM
typedef __attribute__((ext_vector_type(4))) unsigned int u32x4;
typedef __attribute__((ext_vector_type(4))) int i32x4;
typedef __attribute__((ext_vector_type(8))) int i32x8;

// Issue a TDM load of a 2-D bf16 tile (tile0 x tile1, row stride = stride0
// elements) from global memory into LDS at byte offset lds_byte.
// D# packing per CDNA5 ISA (sections 8.3 / 8.4):
//  g0: [1:0]=count=1, [63:32]=lds_addr, [120:64]=global_addr, [127:126]=type=2
//  g1: [17:16]=data_size(1=2B), [79:48]=tensor_dim0, [111:80]=tensor_dim1,
//      [127:112]=tile_dim0, [143:128]=tile_dim1, [207:160]=tensor_dim0_stride
__device__ __forceinline__ void tdm_load_2d(unsigned int lds_byte,
                                            const void* gptr,
                                            unsigned int td0, unsigned int td1,
                                            unsigned int stride0,
                                            unsigned int tile0,
                                            unsigned int tile1) {
  unsigned long long ga = (unsigned long long)gptr;
  u32x4 g0;
  g0[0] = 1u;                                    // count=1 (valid user D#)
  g0[1] = lds_byte;                              // LDS byte address
  g0[2] = (unsigned int)ga;                      // global addr [95:64]
  g0[3] = ((unsigned int)(ga >> 32) & 0x01FFFFFFu) | 0x80000000u;  // type=2
  i32x8 g1;
  g1[0] = (int)(1u << 16);                       // data_size = 1 (2 bytes)
  g1[1] = (int)((td0 & 0xFFFFu) << 16);          // tensor_dim0 lo16 @ [63:48]
  g1[2] = (int)(((td0 >> 16) & 0xFFFFu) | ((td1 & 0xFFFFu) << 16));
  g1[3] = (int)(((td1 >> 16) & 0xFFFFu) | ((tile0 & 0xFFFFu) << 16));
  g1[4] = (int)(tile1 & 0xFFFFu);                // tile_dim1 (tile_dim2 = 0)
  g1[5] = (int)stride0;                          // tensor_dim0_stride lo32
  g1[6] = 0;
  g1[7] = 0;
  i32x4 z4 = {0, 0, 0, 0};
#if defined(__clang_major__) && __clang_major__ >= 23
  i32x8 z8 = {0, 0, 0, 0, 0, 0, 0, 0};
  __builtin_amdgcn_tensor_load_to_lds(g0, g1, z4, z4, z8, 0);
#else
  __builtin_amdgcn_tensor_load_to_lds(g0, g1, z4, z4, 0);
#endif
}
#endif  // HAVE_TDM

// ---------------------------------------------------------------------------
// WMMA helper: D = A(16x32 bf16) x B(32x16 bf16) + C(16x16 f32)
// ---------------------------------------------------------------------------
__device__ __forceinline__ v8f wmma_bf16(v16bf a, v16bf b, v8f c) {
  // (neg_a, A, neg_b, B, c_mod, C, reuse_a, reuse_b)
  return __builtin_amdgcn_wmma_f32_16x16x32_bf16(false, a, false, b, (short)0, c,
                                                 false, false);
}

__device__ __forceinline__ v16bf cat8(v8bf lo, v8bf hi) {
  return __builtin_shufflevector(lo, hi, 0, 1, 2, 3, 4, 5, 6, 7, 8, 9, 10, 11,
                                 12, 13, 14, 15);
}

// Load a 16x32 bf16 fragment (A-layout; identical pattern serves as B-layout
// where "row" = output column n, since both stripe contiguously along K).
// Lane l: row = l&15; K chunks [hi*8 .. hi*8+7] and [16+hi*8 .. 16+hi*8+7].
__device__ __forceinline__ v16bf load_frag(const bf16_t* M, int ld, int row0,
                                           int col0) {
  const int lane = threadIdx.x & 31;
  const bf16_t* p = M + (size_t)(row0 + (lane & 15)) * (size_t)ld +
                    (size_t)(col0 + ((lane >> 4) << 3));
  v8bf lo = *(const v8bf*)p;         // global_load_b128 / ds_load_b128
  v8bf hi = *(const v8bf*)(p + 16);
  return cat8(lo, hi);
}

// ---------------------------------------------------------------------------
// fp32 -> bf16 conversion (vectorized x4)
// ---------------------------------------------------------------------------
__global__ __launch_bounds__(256) void cvt_f32_bf16(const float* __restrict__ in,
                                                    bf16_t* __restrict__ out,
                                                    int n4) {
  int i = blockIdx.x * 256 + threadIdx.x;
  if (i < n4) {
    float4 f = ((const float4*)in)[i];
    v4bf o;
    o[0] = (bf16_t)f.x;
    o[1] = (bf16_t)f.y;
    o[2] = (bf16_t)f.z;
    o[3] = (bf16_t)f.w;
    ((v4bf*)out)[i] = o;
  }
}

// ---------------------------------------------------------------------------
// Projection GEMM: Y[m,n] = sum_k A[m,k] * W[n,k]   (y = A @ W^T)
// A: (4096 x 1024) bf16 row-major, W: (1024 x 1024) bf16 row-major.
// Block = 256 thr = 8 waves arranged 2(m) x 4(n); WG tile 128x256.
// Wave tile 64x64 = 4x4 WMMA tiles: 8 fragment loads -> 16 WMMAs per k-step.
// ---------------------------------------------------------------------------
constexpr int MODE_QK  = 0;  // bf16 -> [b][h][t][d]
constexpr int MODE_VT  = 1;  // bf16 -> [b][h][d][t]  (transposed V per head)
constexpr int MODE_F32 = 2;  // f32  -> [m][n] row-major (final output)

template <int MODE>
__global__ __launch_bounds__(256) void gemm_proj(const bf16_t* __restrict__ A,
                                                 const bf16_t* __restrict__ W,
                                                 void* __restrict__ outp,
                                                 float oscale) {
  const int lane = threadIdx.x & 31;
  const int w = threadIdx.x >> 5;
  const int m0 = blockIdx.y * 128 + (w & 1) * 64;
  const int n0 = blockIdx.x * 256 + (w >> 1) * 64;

  const v8f zero = {0.f, 0.f, 0.f, 0.f, 0.f, 0.f, 0.f, 0.f};
  v8f acc[4][4];
#pragma unroll
  for (int i = 0; i < 4; i++)
#pragma unroll
    for (int j = 0; j < 4; j++) acc[i][j] = zero;

#pragma unroll 2
  for (int k0 = 0; k0 < DM; k0 += 32) {
    if (k0 + 64 < DM) {  // pull next tiles toward the WGP (global_prefetch_b8)
      __builtin_prefetch(A + (size_t)(m0 + (lane & 15)) * DM + k0 + 64, 0, 1);
      __builtin_prefetch(W + (size_t)(n0 + (lane & 15)) * DM + k0 + 64, 0, 1);
    }
    v16bf a[4], b[4];
#pragma unroll
    for (int i = 0; i < 4; i++) a[i] = load_frag(A, DM, m0 + 16 * i, k0);
#pragma unroll
    for (int j = 0; j < 4; j++) b[j] = load_frag(W, DM, n0 + 16 * j, k0);
#pragma unroll
    for (int i = 0; i < 4; i++)
#pragma unroll
      for (int j = 0; j < 4; j++) acc[i][j] = wmma_bf16(a[i], b[j], acc[i][j]);
  }

  const int rbase = (lane >> 4) * 8;
  const int cbase = lane & 15;
#pragma unroll
  for (int i = 0; i < 4; i++)
#pragma unroll
    for (int j = 0; j < 4; j++)
#pragma unroll
      for (int v = 0; v < 8; v++) {
        int m = m0 + 16 * i + rbase + v;
        int n = n0 + 16 * j + cbase;
        float val = acc[i][j][v] * oscale;
        if (MODE == MODE_QK) {
          int b_ = m >> 11, t = m & (TT - 1), h = n >> 6, d = n & (HD - 1);
          ((bf16_t*)outp)[(((size_t)(b_ * NH + h) * TT) + t) * HD + d] =
              (bf16_t)val;
        } else if (MODE == MODE_VT) {
          int b_ = m >> 11, t = m & (TT - 1), h = n >> 6, d = n & (HD - 1);
          ((bf16_t*)outp)[(((size_t)(b_ * NH + h) * HD) + d) * TT + t] =
              (bf16_t)val;
        } else {
          ((float*)outp)[(size_t)m * DM + n] = val;
        }
      }
}

// ---------------------------------------------------------------------------
// FlashAttention forward. Q is pre-scaled by 1/sqrt(hd).
// Grid: (B*H) * (T/128) blocks, 256 threads = 8 waves; wave owns 16 q-rows.
// All 8 waves consume the SAME 64-key K/V chunk, so the chunk is staged into
// LDS once per workgroup by the Tensor Data Mover (double-buffered, issued by
// wave 0, tracked by TENSORcnt, published with a workgroup barrier) while the
// block computes on the previous chunk. Falls back to per-wave global loads
// when the TDM builtin is unavailable.
// ---------------------------------------------------------------------------
__global__ __launch_bounds__(256) void flash_attn(const bf16_t* __restrict__ Q,
                                                  const bf16_t* __restrict__ K,
                                                  const bf16_t* __restrict__ Vt,
                                                  bf16_t* __restrict__ Ao) {
#if HAVE_TDM
  __shared__ __align__(16) bf16_t Kb_s[2][64 * 64];  // [buf][key][d]
  __shared__ __align__(16) bf16_t Vb_s[2][64 * 64];  // [buf][d][key]
#endif
  __shared__ __align__(16) bf16_t p_lds[8][16 * 64];
  const int lane = threadIdx.x & 31;
  const int w = threadIdx.x >> 5;
  const int bh = blockIdx.x >> 4;        // 0..31  (b*16 + h)
  const int qt = blockIdx.x & 15;        // q tile within T
  const int q0 = qt * 128 + w * 16;

  const bf16_t* Qh = Q + (size_t)bh * TT * HD;
  const bf16_t* Kh = K + (size_t)bh * TT * HD;
  const bf16_t* Vh = Vt + (size_t)bh * HD * TT;

  v16bf fq0 = load_frag(Qh, HD, q0, 0);
  v16bf fq1 = load_frag(Qh, HD, q0, 32);

  const v8f zero = {0.f, 0.f, 0.f, 0.f, 0.f, 0.f, 0.f, 0.f};
  v8f o[4];
  float mrow[8], lrow[8];
#pragma unroll
  for (int nt = 0; nt < 4; nt++) o[nt] = zero;
#pragma unroll
  for (int v = 0; v < 8; v++) {
    mrow[v] = -1e30f;
    lrow[v] = 0.f;
  }

  bf16_t* pl = &p_lds[w][0];
  const int rbase = (lane >> 4) * 8;
  const int cbase = lane & 15;

#if HAVE_TDM
  // Prologue: DMA chunk 0 into buffer 0 (one wave issues for the whole WG).
  if (w == 0) {
    tdm_load_2d((unsigned int)(uintptr_t)(void*)&Kb_s[0][0], Kh,
                /*td0=*/HD, /*td1=*/TT, /*stride0=*/HD, /*tile0=*/HD,
                /*tile1=*/64);
    tdm_load_2d((unsigned int)(uintptr_t)(void*)&Vb_s[0][0], Vh,
                /*td0=*/TT, /*td1=*/HD, /*stride0=*/TT, /*tile0=*/64,
                /*tile1=*/HD);
  }
#endif

#pragma unroll 1
  for (int kc = 0; kc < TT; kc += 64) {
#if HAVE_TDM
    const int cur = (kc >> 6) & 1;
    if (w == 0) __builtin_amdgcn_s_wait_tensorcnt(0);  // chunk `cur` landed
    __syncthreads();                                   // publish to all waves
    if (w == 0 && kc + 64 < TT) {                      // overlap next DMA
      tdm_load_2d((unsigned int)(uintptr_t)(void*)&Kb_s[cur ^ 1][0],
                  Kh + (size_t)(kc + 64) * HD, HD, TT, HD, HD, 64);
      tdm_load_2d((unsigned int)(uintptr_t)(void*)&Vb_s[cur ^ 1][0],
                  Vh + (kc + 64), TT, HD, TT, 64, HD);
    }
    const bf16_t* Ksrc = (const bf16_t*)&Kb_s[cur][0];
    const bf16_t* Vsrc = (const bf16_t*)&Vb_s[cur][0];
    const int kbase = 0, vbase = 0, ldk = 64, ldv = 64;
#else
    const bf16_t* Ksrc = Kh;
    const bf16_t* Vsrc = Vh;
    const int kbase = kc, vbase = kc, ldk = HD, ldv = TT;
#endif
    // ---- S = Q @ K^T for 16 q-rows x 64 keys ----
    v8f s[4];
#pragma unroll
    for (int j = 0; j < 4; j++) {
      v16bf fk0 = load_frag(Ksrc, ldk, kbase + 16 * j, 0);
      v16bf fk1 = load_frag(Ksrc, ldk, kbase + 16 * j, 32);
      s[j] = wmma_bf16(fq0, fk0, zero);
      s[j] = wmma_bf16(fq1, fk1, s[j]);
    }
    // ---- online softmax: row stats live per (half-wave, VGPR v) ----
#pragma unroll
    for (int v = 0; v < 8; v++) {
      float mx = fmaxf(fmaxf(s[0][v], s[1][v]), fmaxf(s[2][v], s[3][v]));
      mx = fmaxf(mx, __shfl_xor(mx, 8, 16));
      mx = fmaxf(mx, __shfl_xor(mx, 4, 16));
      mx = fmaxf(mx, __shfl_xor(mx, 2, 16));
      mx = fmaxf(mx, __shfl_xor(mx, 1, 16));
      float mnew = fmaxf(mrow[v], mx);
      float corr = __expf(mrow[v] - mnew);
      mrow[v] = mnew;
      float rs = 0.f;
#pragma unroll
      for (int j = 0; j < 4; j++) {
        float p = __expf(s[j][v] - mnew);
        s[j][v] = p;
        rs += p;
      }
      rs += __shfl_xor(rs, 8, 16);
      rs += __shfl_xor(rs, 4, 16);
      rs += __shfl_xor(rs, 2, 16);
      rs += __shfl_xor(rs, 1, 16);
      lrow[v] = lrow[v] * corr + rs;
#pragma unroll
      for (int nt = 0; nt < 4; nt++) o[nt][v] *= corr;
      // stage P into this wave's private LDS tile (row-major 16x64 bf16)
#pragma unroll
      for (int j = 0; j < 4; j++)
        pl[(rbase + v) * 64 + j * 16 + cbase] = (bf16_t)s[j][v];
    }
    asm volatile("s_wait_dscnt 0x0" ::: "memory");  // LDS writes -> loads
    // ---- O += P @ V (B fragments from d-major V^T are K-contiguous) ----
#pragma unroll
    for (int s2 = 0; s2 < 2; s2++) {
      const bf16_t* ap = pl + (size_t)cbase * 64 + s2 * 32 + ((lane >> 4) << 3);
      v8bf lo = *(const v8bf*)ap;        // ds_load_b128
      v8bf hi = *(const v8bf*)(ap + 16);
      v16bf fp = cat8(lo, hi);
#pragma unroll
      for (int nt = 0; nt < 4; nt++) {
        v16bf fv = load_frag(Vsrc, ldv, nt * 16, vbase + s2 * 32);
        o[nt] = wmma_bf16(fp, fv, o[nt]);
      }
    }
#if HAVE_TDM
    // All reads of buffer `cur` must retire before it is DMA-overwritten at
    // the next iteration's issue point.
    asm volatile("s_wait_dscnt 0x0" ::: "memory");
    __syncthreads();
#endif
  }

  // ---- epilogue: normalize and scatter into [b][t][h*64+d] (bf16) ----
  const int b = bh >> 4, h = bh & (NH - 1);
#pragma unroll
  for (int v = 0; v < 8; v++) {
    float inv = 1.0f / lrow[v];
    int qrow = q0 + rbase + v;
#pragma unroll
    for (int nt = 0; nt < 4; nt++) {
      int d = nt * 16 + cbase;
      Ao[((size_t)b * TT + qrow) * DM + h * HD + d] = (bf16_t)(o[nt][v] * inv);
    }
  }
}

// ---------------------------------------------------------------------------
// Host-side orchestration. Workspace layout (48 MiB):
//   [0,8M)    xb      x in bf16            (4096x1024)
//   [8M,10M)  wqb     W_q bf16             (1024x1024)
//   [10M,12M) wkb     W_k bf16
//   [12M,14M) wvb     W_v bf16
//   [14M,16M) wob     W_o bf16
//   [16M,24M) Qs      Q  [b][h][t][d] bf16 (pre-scaled by 1/8)
//   [24M,32M) Kb      K  [b][h][t][d] bf16
//   [32M,40M) Vtb     V^T [b][h][d][t] bf16
//   [40M,48M) attb    attention out [b][t][D] bf16
// ---------------------------------------------------------------------------
extern "C" void kernel_launch(void* const* d_in, const int* in_sizes, int n_in,
                              void* d_out, int out_size, void* d_ws,
                              size_t ws_size, hipStream_t stream) {
  (void)in_sizes; (void)n_in; (void)out_size; (void)ws_size;
  const float* x  = (const float*)d_in[0];
  const float* Wq = (const float*)d_in[1];
  const float* Wk = (const float*)d_in[2];
  const float* Wv = (const float*)d_in[3];
  const float* Wo = (const float*)d_in[4];
  float* out = (float*)d_out;

  char* ws = (char*)d_ws;
  const size_t MB = 1u << 20;
  bf16_t* xb   = (bf16_t*)(ws + 0 * MB);
  bf16_t* wqb  = (bf16_t*)(ws + 8 * MB);
  bf16_t* wkb  = (bf16_t*)(ws + 10 * MB);
  bf16_t* wvb  = (bf16_t*)(ws + 12 * MB);
  bf16_t* wob  = (bf16_t*)(ws + 14 * MB);
  bf16_t* Qs   = (bf16_t*)(ws + 16 * MB);
  bf16_t* Kb   = (bf16_t*)(ws + 24 * MB);
  bf16_t* Vtb  = (bf16_t*)(ws + 32 * MB);
  bf16_t* attb = (bf16_t*)(ws + 40 * MB);

  // fp32 -> bf16 (x: 4M elems, weights: 1M elems each; x4 vectorized)
  cvt_f32_bf16<<<4096, 256, 0, stream>>>(x, xb, (BB * TT * DM) / 4);
  cvt_f32_bf16<<<1024, 256, 0, stream>>>(Wq, wqb, (DM * DM) / 4);
  cvt_f32_bf16<<<1024, 256, 0, stream>>>(Wk, wkb, (DM * DM) / 4);
  cvt_f32_bf16<<<1024, 256, 0, stream>>>(Wv, wvb, (DM * DM) / 4);
  cvt_f32_bf16<<<1024, 256, 0, stream>>>(Wo, wob, (DM * DM) / 4);

  dim3 g(DM / 256, (BB * TT) / 128);  // (4, 32)
  const float SCALE = 0.125f;         // 1/sqrt(64), folded into Q
  gemm_proj<MODE_QK><<<g, 256, 0, stream>>>(xb, wqb, (void*)Qs, SCALE);
  gemm_proj<MODE_QK><<<g, 256, 0, stream>>>(xb, wkb, (void*)Kb, 1.0f);
  gemm_proj<MODE_VT><<<g, 256, 0, stream>>>(xb, wvb, (void*)Vtb, 1.0f);

  flash_attn<<<BB * NH * (TT / 128), 256, 0, stream>>>(Qs, Kb, Vtb, attb);

  gemm_proj<MODE_F32><<<g, 256, 0, stream>>>(attb, wob, (void*)out, 1.0f);
}
